// ComputeS_45320494907490
// MI455X (gfx1250) — matrix operand order repaired and checked by reading, not compiled, not run on hardware
//
#include <hip/hip_runtime.h>
#include <hip/hip_bf16.h>

// ---------------------------------------------------------------------------
// Problem constants (from reference): B=2, C=64, H=W=96, HW=9216, K=C*9=576
// GEMM per batch: [9216 x 576] * [576 x 9216], fused max/argmax over rows.
// ---------------------------------------------------------------------------
#define BATCH   2
#define CH      64
#define HDIM    96
#define WDIM    96
#define HW      9216            // 96*96
#define KDIM    576             // 64*9
#define KS      18              // KDIM / 32  (k-steps of the 16x16x32 WMMA)
#define NT      576             // HW / 16    (16-wide m/n tiles)
#define FRAG    512             // halfs per 16x32 / 32x16 fragment (32 lanes * 16)
#define EPS     1e-12f

#define HALF_STEPS   9                       // k-steps per async half-tile
#define HALF_BYTES   (HALF_STEPS * FRAG * 2) // 9216 B per half-tile
#define HALF_HALFS   (HALF_STEPS * FRAG)     // 4608 halfs

typedef __attribute__((ext_vector_type(16))) _Float16 v16h;
typedef __attribute__((ext_vector_type(8)))  float    v8f;

// workspace layout (bytes)
#define AWS_HALFS ((size_t)BATCH * NT * KS * FRAG)      // 10,616,832 halfs
#define AWS_BYTES (AWS_HALFS * 2)                       // 21,233,664 B

// ---------------------------------------------------------------------------
// k0: per-pixel channel sum-of-squares for both tensors
// ---------------------------------------------------------------------------
__global__ __launch_bounds__(256) void qsum_kernel(const float* __restrict__ refT,
                                                   const float* __restrict__ lrT,
                                                   float* __restrict__ qref,
                                                   float* __restrict__ qlr) {
    int idx = blockIdx.x * 256 + threadIdx.x;           // (b, p)
    if (idx >= BATCH * HW) return;
    int b = idx / HW, p = idx - b * HW;
    const float* r = refT + (size_t)b * CH * HW + p;
    const float* l = lrT  + (size_t)b * CH * HW + p;
    float sr = 0.f, sl = 0.f;
    for (int c = 0; c < CH; ++c) {
        float a = r[(size_t)c * HW]; sr += a * a;
        float d = l[(size_t)c * HW]; sl += d * d;
    }
    qref[idx] = sr; qlr[idx] = sl;
}

// ---------------------------------------------------------------------------
// k1: per-location patch inverse norms: 1 / max(sqrt(sum 3x3 of q), eps)
// ---------------------------------------------------------------------------
__global__ __launch_bounds__(256) void invnorm_kernel(const float* __restrict__ qref,
                                                      const float* __restrict__ qlr,
                                                      float* __restrict__ invA,
                                                      float* __restrict__ invB) {
    int idx = blockIdx.x * 256 + threadIdx.x;
    if (idx >= BATCH * HW) return;
    int b = idx / HW, p = idx - b * HW;
    int hh = p / WDIM, ww = p - hh * WDIM;
    float sr = 0.f, sl = 0.f;
    #pragma unroll
    for (int di = -1; di <= 1; ++di)
        #pragma unroll
        for (int dj = -1; dj <= 1; ++dj) {
            int y = hh + di, x = ww + dj;
            if ((unsigned)y < (unsigned)HDIM && (unsigned)x < (unsigned)WDIM) {
                int q = b * HW + y * WDIM + x;
                sr += qref[q]; sl += qlr[q];
            }
        }
    invA[idx] = 1.f / fmaxf(sqrtf(sr), EPS);
    invB[idx] = 1.f / fmaxf(sqrtf(sl), EPS);
}

// ---------------------------------------------------------------------------
// k2: scatter normalized f16 patch values into WMMA fragment order.
//   A (16-bit 16x32, ISA 7.12.2): lane = 16*hA + row; kk = 8*hA + e (+8 if e>=8)
//   B (16-bit 32x16):             lane = 16*hB + col; kk = 8*hB + e (+8 if e>=8)
// ---------------------------------------------------------------------------
__global__ __launch_bounds__(256) void scatter_kernel(const float* __restrict__ refT,
                                                      const float* __restrict__ lrT,
                                                      const float* __restrict__ invA,
                                                      const float* __restrict__ invB,
                                                      _Float16* __restrict__ Aws,
                                                      _Float16* __restrict__ Bws) {
    int idx = blockIdx.x * 256 + threadIdx.x;           // (b*HW + p)*8 + g
    if (idx >= BATCH * HW * 8) return;
    int g  = idx & 7;
    int bp = idx >> 3;
    int b  = bp / HW, p = bp - b * HW;
    int hh = p / WDIM, ww = p - hh * WDIM;
    int T  = p >> 4;                                    // tile id (rows for A, cols for B)
    int r  = p & 15;                                    // row/col within tile
    float sA = invA[bp], sB = invB[bp];
    size_t base = ((size_t)(b * NT + T)) * KS * FRAG;

    #pragma unroll
    for (int cc = 0; cc < 8; ++cc) {
        int c = g * 8 + cc;
        #pragma unroll
        for (int t = 0; t < 9; ++t) {
            int di = t / 3 - 1, dj = t % 3 - 1;
            int y = hh + di, x = ww + dj;
            float vr = 0.f, vl = 0.f;
            if ((unsigned)y < (unsigned)HDIM && (unsigned)x < (unsigned)WDIM) {
                size_t off = ((size_t)b * CH + c) * HW + (size_t)y * WDIM + x;
                vr = refT[off]; vl = lrT[off];
            }
            int k  = c * 9 + t;                         // unfold index = c*9 + tap
            int s  = k >> 5, kk = k & 31;
            int hA = (kk >> 3) & 1;
            int eA = (kk & 7) + ((kk >= 16) ? 8 : 0);
            Aws[base + (size_t)s * FRAG + (16 * hA + r) * 16 + eA] = (_Float16)(vr * sA);
            int hB = (kk >> 3) & 1;
            int eB = (kk & 7) + ((kk >= 16) ? 8 : 0);
            Bws[base + (size_t)s * FRAG + (16 * hB + r) * 16 + eB] = (_Float16)(vl * sB);
        }
    }
}

// ---------------------------------------------------------------------------
// k3: fused WMMA GEMM + argmax with CDNA5 async global->LDS A staging.
// One workgroup = 8 waves, 2 n-tiles (B register-resident, 288 VGPRs/wave).
// Each wave sweeps 72 m-tiles; A is streamed by the async engine into a
// private per-wave LDS double buffer at half-m-tile (9 k-step) granularity:
// 18x global_load_async_to_lds_b128 (ASYNCcnt) cover ~18 WMMAs of compute,
// so L2 latency is hidden without VGPR cost; inner loop reads A via ds_load.
// ---------------------------------------------------------------------------
__device__ __forceinline__ v8f wmma_f16(v16h a, v16h b, v8f c) {
    return __builtin_amdgcn_wmma_f32_16x16x32_f16(false, a, false, b,
                                                  (short)0, c, false, false);
}

// copy one half-tile (9216 B) global -> LDS via 18 async b128 ops.
// IOFFSET is added to BOTH the LDS and global addresses (ISA 15.18.3),
// so fixed base VGPRs + stepped immediate produce a linear copy.
__device__ __forceinline__ void async_copy_half(unsigned lds32, unsigned long long gaddr) {
    #pragma unroll
    for (int i = 0; i < 18; ++i) {
        asm volatile("global_load_async_to_lds_b128 %0, %1, off offset:%c2"
                     :: "v"(lds32), "v"(gaddr), "i"(i * 512)
                     : "memory");
    }
}

__global__ __launch_bounds__(256) void gemm_argmax_kernel(const _Float16* __restrict__ Aws,
                                                          const _Float16* __restrict__ Bws,
                                                          float* __restrict__ outS,
                                                          float* __restrict__ outH) {
    __shared__ _Float16 Ash[8 * 2 * HALF_HALFS];        // 8 waves x 2 bufs x 9 KB = 144 KB
    __shared__ float redV[8][32];
    __shared__ int   redI[8][32];

    const int b    = blockIdx.y;
    const int nt0  = blockIdx.x * 2;                    // first of 2 n-tiles
    const int wave = threadIdx.x >> 5;
    const int lane = threadIdx.x & 31;
    const int half = lane >> 4;                         // C layout: rows r / r+8

    // LDS byte address of this wave's slice (+ this lane's 16B slot)
    const unsigned sliceB =
        (unsigned)(unsigned long long)(const void*)&Ash[0]
        + (unsigned)wave * (2 * HALF_BYTES) + (unsigned)lane * 16;

    // kick off the async engine on this wave's first half-tile immediately
    {
        unsigned long long g = (unsigned long long)
            ((const char*)Aws + ((size_t)(b * NT + wave)) * KS * FRAG * 2 + lane * 16);
        async_copy_half(sliceB, g);
    }

    // B fragments for both n-tiles: register-resident (loaded once from global)
    v16h Bq0[KS], Bq1[KS];
    {
        const _Float16* B0 = Bws + ((size_t)(b * NT + nt0)) * KS * FRAG + lane * 16;
        const _Float16* B1 = B0 + (size_t)KS * FRAG;
        #pragma unroll
        for (int s = 0; s < KS; ++s) {
            Bq0[s] = *(const v16h*)(B0 + (size_t)s * FRAG);
            Bq1[s] = *(const v16h*)(B1 + (size_t)s * FRAG);
        }
    }

    float bv0 = -3.0e38f, bv1 = -3.0e38f;               // running best per n-tile
    int   bi0 = 0,        bi1 = 0;

    for (int ti = 0; ti < NT / 8; ++ti) {               // 72 m-tiles per wave
        const int T = wave + ti * 8;
        v8f acc0 = {}, acc1 = {};
        #pragma unroll
        for (int hh = 0; hh < 2; ++hh) {                // two async half-tiles per m-tile
            int ht = ti * 2 + hh;
            if (ht + 1 < 2 * (NT / 8)) {
                // issue next half-tile into the other buffer, then wait for current
                int Tn = wave + ((ht + 1) >> 1) * 8;
                int hn = (ht + 1) & 1;                  // half index within tile Tn
                unsigned long long g = (unsigned long long)
                    ((const char*)Aws +
                     (((size_t)(b * NT + Tn)) * KS + (size_t)hn * HALF_STEPS) * FRAG * 2 +
                     lane * 16);
                async_copy_half(sliceB + (unsigned)((ht + 1) & 1) * HALF_BYTES, g);
                asm volatile("s_wait_asynccnt 18" ::: "memory");   // older 18 (current buf) done
            } else {
                asm volatile("s_wait_asynccnt 0" ::: "memory");    // drain for last buffer
            }
            const _Float16* Ab = Ash + (size_t)wave * (2 * HALF_HALFS)
                                     + (size_t)(ht & 1) * HALF_HALFS + lane * 16;
            #pragma unroll
            for (int s = 0; s < HALF_STEPS; ++s) {
                v16h a = *(const v16h*)(Ab + (size_t)s * FRAG);    // ds_load_b128 x2
                acc0 = wmma_f16(a, Bq0[hh * HALF_STEPS + s], acc0);
                acc1 = wmma_f16(a, Bq1[hh * HALF_STEPS + s], acc1);
            }
        }

        int mb = T * 16 + 8 * half;                     // rows of m-tile T in this lane
        #pragma unroll
        for (int rr = 0; rr < 8; ++rr) {
            int m = mb + rr;
            float v = acc0[rr];
            if (v > bv0 || (v == bv0 && m < bi0)) { bv0 = v; bi0 = m; }
            v = acc1[rr];
            if (v > bv1 || (v == bv1 && m < bi1)) { bv1 = v; bi1 = m; }
        }
    }

    // combine the two row-halves (lane ^ 16 holds same column, other 8 rows)
    {
        float ov = __shfl_xor(bv0, 16, 32); int oi = __shfl_xor(bi0, 16, 32);
        if (ov > bv0 || (ov == bv0 && oi < bi0)) { bv0 = ov; bi0 = oi; }
        ov = __shfl_xor(bv1, 16, 32); oi = __shfl_xor(bi1, 16, 32);
        if (ov > bv1 || (ov == bv1 && oi < bi1)) { bv1 = ov; bi1 = oi; }
    }
    if (lane < 16) {
        redV[wave][lane]      = bv0; redI[wave][lane]      = bi0;
        redV[wave][16 + lane] = bv1; redI[wave][16 + lane] = bi1;
    }
    __syncthreads();

    // cross-wave reduction + write: 32 columns per workgroup
    if (threadIdx.x < 32) {
        float bv = redV[0][threadIdx.x]; int bi = redI[0][threadIdx.x];
        #pragma unroll
        for (int w2 = 1; w2 < 8; ++w2) {
            float v = redV[w2][threadIdx.x]; int i2 = redI[w2][threadIdx.x];
            if (v > bv || (v == bv && i2 < bi)) { bv = v; bi = i2; }
        }
        int n = nt0 * 16 + threadIdx.x;                 // column index within batch
        outS[(size_t)b * HW + n] = bv;
        outH[(size_t)b * HW + n] = (float)bi;
    }
}

// ---------------------------------------------------------------------------
// launch: inputs are {lrsr_lv2, ref_lv2} (dict order).  lr -> B (columns n),
// ref -> A (rows m, reduced by argmax).  Output = S (18432 f32) ++ Hmap (18432).
// ---------------------------------------------------------------------------
extern "C" void kernel_launch(void* const* d_in, const int* in_sizes, int n_in,
                              void* d_out, int out_size, void* d_ws, size_t ws_size,
                              hipStream_t stream) {
    const float* lrT  = (const float*)d_in[0];          // lrsr_lv2
    const float* refT = (const float*)d_in[1];          // ref_lv2

    float* outS = (float*)d_out;
    float* outH = outS + BATCH * HW;

    char* ws = (char*)d_ws;
    _Float16* Aws = (_Float16*)ws;                      // ref fragments  (21.2 MB)
    _Float16* Bws = (_Float16*)(ws + AWS_BYTES);        // lr  fragments  (21.2 MB)
    float* qref = (float*)(ws + 2 * AWS_BYTES);
    float* qlr  = qref + BATCH * HW;
    float* invA = qlr  + BATCH * HW;
    float* invB = invA + BATCH * HW;

    qsum_kernel<<<(BATCH * HW + 255) / 256, 256, 0, stream>>>(refT, lrT, qref, qlr);
    invnorm_kernel<<<(BATCH * HW + 255) / 256, 256, 0, stream>>>(qref, qlr, invA, invB);
    scatter_kernel<<<(BATCH * HW * 8 + 255) / 256, 256, 0, stream>>>(refT, lrT, invA, invB, Aws, Bws);

    dim3 grid(NT / 2, BATCH);                           // 288 x 2 workgroups
    gemm_argmax_kernel<<<grid, 256, 0, stream>>>(Aws, Bws, outS, outH);
}
